// MultiHeadGATLayer_86071144612506
// MI455X (gfx1250) — matrix operand order repaired and checked by reading, not compiled, not run on hardware
//
#include <hip/hip_runtime.h>
#include <hip/hip_bf16.h>

// ---------------------------------------------------------------------------
// MultiHeadGAT on gfx1250:
//   z = h @ W    via v_wmma_f32_16x16x32_bf16 with bf16 hi/lo split (3 WMMAs
//                per 32-wide K step -> ~fp32 accuracy on the fast path)
//   segment softmax + scatter-add via native f32 atomics (L2-resident)
// ---------------------------------------------------------------------------

typedef __attribute__((ext_vector_type(16))) __bf16 v16bf;
typedef __attribute__((ext_vector_type(8)))  float  v8f;

#define WMMA_BF16(A, B, C) \
  __builtin_amdgcn_wmma_f32_16x16x32_bf16(false, (A), false, (B), (short)0, (C), false, false)

#define IN_DIM  256
#define NCOL    256   // N_HEADS * OUT_DIM
#define N_HEADS 4
#define OUT_DIM 64

// ---------------------------------------------------------------------------
// Fast f32 atomics
// ---------------------------------------------------------------------------
__device__ __forceinline__ void atomAddF(float* p, float v) {
#if defined(__HIP_PLATFORM_AMD__) || defined(__HIP_DEVICE_COMPILE__)
  unsafeAtomicAdd(p, v);          // -> global_atomic_add_f32 (no CAS loop)
#else
  atomicAdd(p, v);
#endif
}

// float atomic-max via signed/unsigned integer ordering trick.
// Works with a single array initialized to -inf (0xFF800000).
__device__ __forceinline__ void atomMaxF(float* p, float v) {
  if (v >= 0.0f) atomicMax((int*)p, __float_as_int(v));
  else           atomicMin((unsigned int*)p, __float_as_uint(v));
}

// ---------------------------------------------------------------------------
// Kernel P: repack W[h][i][o] (fp32) into WMMA-B fragment layout, bf16 hi/lo.
// B is the [K=256][N=256] matrix with column n = h*64+o.
// Fragment layout (16-bit B, 32x16, wave32): lane n=lane&15,
//   element e -> K = ktile*32 + (lane>>4)*16 + e.  Each lane's 16 bf16 are
//   contiguous (32B) so the GEMM does two b128 loads per fragment.
// Index = ((ktile*16 + ntile)*32 + lane)*16 + e ; total 65536 elements.
// ---------------------------------------------------------------------------
__global__ __launch_bounds__(256) void gat_pack_w(const float* __restrict__ W,
                                                  __bf16* __restrict__ Bhi,
                                                  __bf16* __restrict__ Blo) {
  int idx  = blockIdx.x * 256 + threadIdx.x;       // < 65536
  int e    = idx & 15;
  int lane = (idx >> 4) & 31;
  int nt   = (idx >> 9) & 15;
  int kt   = idx >> 13;
  int n    = nt * 16 + (lane & 15);
  int K    = kt * 32 + (lane >> 4) * 16 + e;
  float val = W[(size_t)(n >> 6) * (IN_DIM * OUT_DIM) + (size_t)K * OUT_DIM + (n & 63)];
  __bf16 hi = (__bf16)val;
  Bhi[idx] = hi;
  Blo[idx] = (__bf16)(val - (float)hi);
}

// ---------------------------------------------------------------------------
// Init: out = 0, m = -inf, denom = 0
// ---------------------------------------------------------------------------
__global__ __launch_bounds__(256) void gat_init(float* __restrict__ out,
                                                float* __restrict__ mmax,
                                                float* __restrict__ denom,
                                                int n_nodes) {
  int i = blockIdx.x * 256 + threadIdx.x;
  if (i < n_nodes * NCOL) out[i] = 0.0f;
  if (i < n_nodes * N_HEADS) { mmax[i] = -__builtin_inff(); denom[i] = 0.0f; }
}

// ---------------------------------------------------------------------------
// Projection GEMM: one block = 16 node rows x 256 cols. 8 waves; wave w owns
// column tiles 2w and 2w+1. A tile staged in LDS pre-packed into a contiguous-k
// 16-bit A-fragment layout (lane m = lane&15; element e:
//   K = (lane>>4)*16 + e ), bf16 hi/lo.
// ---------------------------------------------------------------------------
__global__ __launch_bounds__(256) void gat_proj(const float* __restrict__ h,
                                                const __bf16* __restrict__ Bhi,
                                                const __bf16* __restrict__ Blo,
                                                float* __restrict__ z,
                                                int n_nodes) {
  __shared__ __align__(32) __bf16 Ahi[8][32][16];   // 8 KB
  __shared__ __align__(32) __bf16 Alo[8][32][16];   // 8 KB

  const int t    = threadIdx.x;
  const int lane = t & 31;
  const int wave = t >> 5;
  const int row0 = blockIdx.x * 16;

  // --- stage + split A tile: thread t fills (ktile = wave, lane) fragment row
  {
    const int kt = wave;                 // 0..7 (one 32-wide K slab per wave)
    const int m  = lane & 15;
    const int lh = lane >> 4;
    const int row = row0 + m;
    const float* hrow = h + (size_t)row * IN_DIM;
#pragma unroll
    for (int e = 0; e < 16; ++e) {
      int v = e >> 1, p = e & 1;
      int K = kt * 32 + lh * 16 + e; (void)v; (void)p;
      float val = (row < n_nodes) ? hrow[K] : 0.0f;
      __bf16 hi = (__bf16)val;
      Ahi[kt][lane][e] = hi;
      Alo[kt][lane][e] = (__bf16)(val - (float)hi);
    }
  }
  __syncthreads();

  const int nt0 = wave * 2;
  const int nt1 = wave * 2 + 1;
  const v16bf* bh = (const v16bf*)Bhi;
  const v16bf* bl = (const v16bf*)Blo;

  v8f acc0 = {};
  v8f acc1 = {};

#pragma unroll
  for (int k = 0; k < 8; ++k) {
    v16bf ahi = *(const v16bf*)(&Ahi[k][lane][0]);
    v16bf alo = *(const v16bf*)(&Alo[k][lane][0]);
    v16bf b0h = bh[(size_t)(k * 16 + nt0) * 32 + lane];
    v16bf b0l = bl[(size_t)(k * 16 + nt0) * 32 + lane];
    v16bf b1h = bh[(size_t)(k * 16 + nt1) * 32 + lane];
    v16bf b1l = bl[(size_t)(k * 16 + nt1) * 32 + lane];
    // bf16 split-precision: hi*hi + lo*hi + hi*lo
    acc0 = WMMA_BF16(ahi, b0h, acc0);
    acc0 = WMMA_BF16(alo, b0h, acc0);
    acc0 = WMMA_BF16(ahi, b0l, acc0);
    acc1 = WMMA_BF16(ahi, b1h, acc1);
    acc1 = WMMA_BF16(alo, b1h, acc1);
    acc1 = WMMA_BF16(ahi, b1l, acc1);
  }

  // C/D layout: VGPR r, lane l -> (M = r + (l>>4)*8, N = l&15)
  const int col0  = nt0 * 16 + (lane & 15);
  const int col1  = nt1 * 16 + (lane & 15);
  const int rbase = row0 + (lane >> 4) * 8;
#pragma unroll
  for (int r = 0; r < 8; ++r) {
    int rr = rbase + r;
    if (rr < n_nodes) {
      z[(size_t)rr * NCOL + col0] = acc0[r];
      z[(size_t)rr * NCOL + col1] = acc1[r];
    }
  }
}

// ---------------------------------------------------------------------------
// Per-(node, head) attention coefficients: e_src = z.a_src, e_dst = z.a_dst
// ---------------------------------------------------------------------------
__global__ __launch_bounds__(256) void gat_attn(const float* __restrict__ z,
                                                const float* __restrict__ a_src,
                                                const float* __restrict__ a_dst,
                                                float* __restrict__ es,
                                                float* __restrict__ ed,
                                                int n_nodes) {
  int i = blockIdx.x * 256 + threadIdx.x;
  if (i >= n_nodes * N_HEADS) return;
  int node = i >> 2, hd = i & 3;
  const float* zr = z + (size_t)node * NCOL + hd * OUT_DIM;
  const float* as = a_src + hd * OUT_DIM;
  const float* ad = a_dst + hd * OUT_DIM;
  float s = 0.0f, d = 0.0f;
#pragma unroll 8
  for (int o = 0; o < OUT_DIM; ++o) { s += zr[o] * as[o]; d += zr[o] * ad[o]; }
  es[i] = s;
  ed[i] = d;
}

// ---------------------------------------------------------------------------
// Edge logits: leaky_relu(e_src[src] + e_dst[dst]) + segment atomic max
// ---------------------------------------------------------------------------
__global__ __launch_bounds__(256) void gat_edge_logits(const float* __restrict__ es,
                                                       const float* __restrict__ ed,
                                                       const int* __restrict__ src,
                                                       const int* __restrict__ dst,
                                                       float* __restrict__ e_ws,
                                                       float* __restrict__ mmax,
                                                       int n_edges) {
  int i = blockIdx.x * 256 + threadIdx.x;
  if (i >= n_edges * N_HEADS) return;
  int eid = i >> 2, hd = i & 3;
  int s = src[eid], d = dst[eid];
  float v = es[s * N_HEADS + hd] + ed[d * N_HEADS + hd];
  v = (v > 0.0f) ? v : 0.01f * v;          // leaky_relu, slope 0.01
  e_ws[i] = v;
  atomMaxF(&mmax[d * N_HEADS + hd], v);
}

// ---------------------------------------------------------------------------
// exp(e - m[dst]) + segment atomic sum
// ---------------------------------------------------------------------------
__global__ __launch_bounds__(256) void gat_edge_exp(const int* __restrict__ dst,
                                                    const float* __restrict__ mmax,
                                                    float* __restrict__ e_ws,
                                                    float* __restrict__ denom,
                                                    int n_edges) {
  int i = blockIdx.x * 256 + threadIdx.x;
  if (i >= n_edges * N_HEADS) return;
  int eid = i >> 2, hd = i & 3;
  int d = dst[eid];
  float ex = expf(e_ws[i] - mmax[d * N_HEADS + hd]);
  e_ws[i] = ex;
  atomAddF(&denom[d * N_HEADS + hd], ex);
}

// ---------------------------------------------------------------------------
// Weighted scatter-sum: block per edge, thread per output channel (coalesced
// z[src] gather + coalesced f32 atomic row-add into out[dst]).
// ---------------------------------------------------------------------------
__global__ __launch_bounds__(256) void gat_aggregate(const float* __restrict__ z,
                                                     const float* __restrict__ e_ws,
                                                     const float* __restrict__ denom,
                                                     const int* __restrict__ src,
                                                     const int* __restrict__ dst,
                                                     float* __restrict__ out,
                                                     int n_edges) {
  int eid = blockIdx.x;
  int c   = threadIdx.x;                  // 0..255
  int s = src[eid], d = dst[eid];
  int hd = c >> 6;                        // channel -> head
  float alpha = e_ws[eid * N_HEADS + hd] / denom[d * N_HEADS + hd];
  float val = alpha * z[(size_t)s * NCOL + c];
  atomAddF(&out[(size_t)d * NCOL + c], val);
}

// ---------------------------------------------------------------------------
// Launcher
// ---------------------------------------------------------------------------
extern "C" void kernel_launch(void* const* d_in, const int* in_sizes, int n_in,
                              void* d_out, int out_size, void* d_ws, size_t ws_size,
                              hipStream_t stream) {
  const float* h     = (const float*)d_in[0];
  const int*   src   = (const int*)d_in[1];
  const int*   dst   = (const int*)d_in[2];
  const float* W     = (const float*)d_in[3];
  const float* a_src = (const float*)d_in[4];
  const float* a_dst = (const float*)d_in[5];
  float* out = (float*)d_out;

  const int n_nodes = in_sizes[0] / IN_DIM;
  const int n_edges = in_sizes[1];

  // workspace layout (all offsets 16B-aligned by construction)
  char* ws = (char*)d_ws;
  size_t off = 0;
  __bf16* Bhi = (__bf16*)(ws + off); off += (size_t)65536 * 2;
  __bf16* Blo = (__bf16*)(ws + off); off += (size_t)65536 * 2;
  float* z    = (float*)(ws + off);  off += (size_t)n_nodes * NCOL * 4;
  float* e_ws = (float*)(ws + off);  off += (size_t)n_edges * N_HEADS * 4;
  float* mmax = (float*)(ws + off);  off += (size_t)n_nodes * N_HEADS * 4;
  float* den  = (float*)(ws + off);  off += (size_t)n_nodes * N_HEADS * 4;
  float* es   = (float*)(ws + off);  off += (size_t)n_nodes * N_HEADS * 4;
  float* ed   = (float*)(ws + off);  off += (size_t)n_nodes * N_HEADS * 4;
  (void)ws_size; (void)n_in; (void)out_size;

  gat_pack_w<<<256, 256, 0, stream>>>(W, Bhi, Blo);
  gat_init<<<(n_nodes * NCOL + 255) / 256, 256, 0, stream>>>(out, mmax, den, n_nodes);
  gat_proj<<<(n_nodes + 15) / 16, 256, 0, stream>>>(h, Bhi, Blo, z, n_nodes);
  gat_attn<<<(n_nodes * N_HEADS + 255) / 256, 256, 0, stream>>>(z, a_src, a_dst, es, ed, n_nodes);
  gat_edge_logits<<<(n_edges * N_HEADS + 255) / 256, 256, 0, stream>>>(es, ed, src, dst, e_ws, mmax, n_edges);
  gat_edge_exp<<<(n_edges * N_HEADS + 255) / 256, 256, 0, stream>>>(dst, mmax, e_ws, den, n_edges);
  gat_aggregate<<<n_edges, 256, 0, stream>>>(z, e_ws, den, src, dst, out, n_edges);
}
